// MultiHeadAttV2_34325378629733
// MI455X (gfx1250) — matrix operand order repaired and checked by reading, not compiled, not run on hardware
//
#include <hip/hip_runtime.h>
#include <hip/hip_bf16.h>
#include <math.h>
#include <stdint.h>

typedef __bf16 bf16;
typedef __attribute__((ext_vector_type(16))) __bf16 v16bf;
typedef __attribute__((ext_vector_type(8)))  __bf16 v8bf;
typedef __attribute__((ext_vector_type(8)))  float  v8f;

#define BSZ 1024
#define LSZ 200
#define LP  208
#define DQq 512
#define DFf 128
#define HH  4
#define DD1 128
#define DD2 64
#define STRD 136   // padded LDS row stride (elements)

// workspace byte offsets
#define WS_FQV   0u
#define WS_HIDQ  1048576u
#define WS_WF1T  3145728u
#define WS_WV1T  3276800u
#define WS_WF2T  3407872u
#define WS_WV2T  3473408u

// LDS byte offsets (dynamic shared memory carve)
#define SM_FACT  0u
#define SM_HID   56576u
#define SM_W     113152u
#define SM_B     147968u
#define SM_DOT   148480u
#define SM_RED   149312u
#define SM_OUT   149376u
#define SM_TOTAL 150400u

__device__ __forceinline__ float silu_f(float x) { return x / (1.f + __expf(-x)); }

__device__ __forceinline__ v16bf pack_frag(const bf16* p0, const bf16* p1) {
    v8bf c0 = *(const v8bf*)p0;
    v8bf c1 = *(const v8bf*)p1;
    v16bf r;
#pragma unroll
    for (int i = 0; i < 8; ++i) { r[i] = c0[i]; r[i + 8] = c1[i]; }
    return r;
}

// A fragment: 16x32 bf16 tile, rows mBase..mBase+15, cols kBase..kBase+31
__device__ __forceinline__ v16bf load_a(const bf16* base, int mBase, int kBase, int lane) {
    int row = mBase + (lane & 15);
    int sh  = (lane < 16) ? 0 : 8;
    const bf16* p = base + row * STRD + kBase + sh;
    return pack_frag(p, p + 16);
}

// B fragment from transposed weights Wt[N][K]: cols nBase..+15, K rows kBase..+31
__device__ __forceinline__ v16bf load_b(const bf16* base, int nBase, int kBase, int lane) {
    int row = nBase + (lane & 15);
    int sh  = (lane < 16) ? 0 : 8;
    const bf16* p = base + row * STRD + kBase + sh;
    return pack_frag(p, p + 16);
}

__device__ __forceinline__ v8f wmma_bf16(v16bf a, v16bf b, v8f c) {
    return __builtin_amdgcn_wmma_f32_16x16x32_bf16(false, a, false, b, (short)0, c, false, false);
}

// Async-stage `rows`x128 bf16 weight matrix from global (packed) into LDS with
// padded stride, 16B per lane per issue, tracked by ASYNCcnt (CDNA5 async path).
__device__ __forceinline__ void stage_weights_async(const bf16* gsrc, unsigned ldsByteBase,
                                                    int rows, int tid) {
    const unsigned long long gbase = (unsigned long long)(uintptr_t)gsrc;
    const int nchunks = rows * 16;  // 16-byte chunks per 128-elem row
    for (int c = tid; c < nchunks; c += 256) {
        const int r = c >> 4, seg = c & 15;
        unsigned lds  = ldsByteBase + (unsigned)(r * (STRD * 2) + seg * 16);
        unsigned goff = (unsigned)(c * 16);
        asm volatile("global_load_async_to_lds_b128 %0, %1, %2"
                     :: "v"(lds), "v"(goff), "s"(gbase) : "memory");
    }
}

__device__ __forceinline__ void wait_async_all() {
    asm volatile("s_wait_asynccnt 0" ::: "memory");
}

// hid = A(sA LPxK128) x W(sW Nx128) + bias -> sHid bf16, N = NT*16 columns
template <int NT>
__device__ __forceinline__ void gemm_store_hid(const bf16* sA, const bf16* sW, const float* sB,
                                               bf16* sHid, int wid, int lane) {
    const int lm = lane & 15;
    const int hiOff = (lane < 16) ? 0 : 8;
    for (int mt = wid; mt < 13; mt += 8) {
        const int mBase = mt * 16;
        v8f acc[NT];
#pragma unroll
        for (int nt = 0; nt < NT; ++nt)
#pragma unroll
            for (int r = 0; r < 8; ++r) acc[nt][r] = 0.f;
#pragma unroll
        for (int ks = 0; ks < 4; ++ks) {
            v16bf a = load_a(sA, mBase, ks * 32, lane);
#pragma unroll
            for (int nt = 0; nt < NT; ++nt) {
                v16bf bb = load_b(sW, nt * 16, ks * 32, lane);
                acc[nt] = wmma_bf16(a, bb, acc[nt]);
            }
        }
#pragma unroll
        for (int nt = 0; nt < NT; ++nt) {
            const int col = nt * 16 + lm;
            const float bias = sB[col];
#pragma unroll
            for (int r = 0; r < 8; ++r) {
                const int row = mBase + r + hiOff;
                sHid[row * STRD + col] = (bf16)(acc[nt][r] + bias);
            }
        }
    }
}

// ---------------- prologue kernels ----------------

// W [H][IN][OUT] f32 -> Wt [H][OUT][IN] bf16
__global__ void k_transpose(const float* __restrict__ W, bf16* __restrict__ Wt, int IN, int OUT) {
    int total = HH * IN * OUT;
    for (int idx = blockIdx.x * blockDim.x + threadIdx.x; idx < total; idx += gridDim.x * blockDim.x) {
        int h = idx / (IN * OUT);
        int rem = idx % (IN * OUT);
        int o = rem / IN, i = rem % IN;
        Wt[idx] = (bf16)W[(h * IN + i) * OUT + o];
    }
}

// hidQ[h][b][o] = query[b]·Wq1[h][:,o] + bq1
__global__ void k_hidq(const float* __restrict__ q, const float* __restrict__ Wq1,
                       const float* __restrict__ bq1, float* __restrict__ hidQ) {
    int idx = blockIdx.x * blockDim.x + threadIdx.x;
    if (idx >= HH * BSZ * DD1) return;
    int h = idx / (BSZ * DD1);
    int rem = idx % (BSZ * DD1);
    int b = rem / DD1, o = rem % DD1;
    float s = bq1[h * DD1 + o];
    const float* qr = q + (size_t)b * DQq;
    const float* w  = Wq1 + (size_t)h * DQq * DD1 + o;
    for (int i = 0; i < DQq; ++i) s += qr[i] * w[i * DD1];
    hidQ[idx] = s;
}

// fqv[h][b][o] = silu(hidQ[h][b]·Wq2[h][:,o] + bq2)
__global__ void k_fq(const float* __restrict__ hidQ, const float* __restrict__ Wq2,
                     const float* __restrict__ bq2, float* __restrict__ fqv) {
    int idx = blockIdx.x * blockDim.x + threadIdx.x;
    if (idx >= HH * BSZ * DD2) return;
    int h = idx / (BSZ * DD2);
    int rem = idx % (BSZ * DD2);
    int b = rem / DD2, o = rem % DD2;
    float s = bq2[h * DD2 + o];
    const float* hr = hidQ + ((size_t)h * BSZ + b) * DD1;
    const float* w  = Wq2 + (size_t)h * DD1 * DD2 + o;
    for (int i = 0; i < DD1; ++i) s += hr[i] * w[i * DD2];
    fqv[idx] = silu_f(s);
}

// ---------------- main attention kernel: one workgroup per batch b ----------------

__global__ __launch_bounds__(256) void k_attn(
    const float* __restrict__ fact, const int* __restrict__ mask, const float* __restrict__ mm,
    const float* __restrict__ bf1, const float* __restrict__ bf2,
    const float* __restrict__ bv1, const float* __restrict__ bv2,
    const float* __restrict__ tau1, const float* __restrict__ tau2,
    const float* __restrict__ fqv,
    const bf16* __restrict__ Wf1t, const bf16* __restrict__ Wf2t,
    const bf16* __restrict__ Wv1t, const bf16* __restrict__ Wv2t,
    float* __restrict__ out) {
    extern __shared__ char smem[];
    bf16*  sFact = (bf16*)(smem + SM_FACT);   // [LP][STRD]
    bf16*  sHid  = (bf16*)(smem + SM_HID);    // [LP][STRD]
    bf16*  sW    = (bf16*)(smem + SM_W);      // [128][STRD]
    float* sB    = (float*)(smem + SM_B);     // [128]
    float* sDot  = (float*)(smem + SM_DOT);   // [LP]  (dot, then alpha)
    float* sRed  = (float*)(smem + SM_RED);   // [16]
    float* sOut  = (float*)(smem + SM_OUT);   // [256]

    const unsigned sWlds = (unsigned)(uintptr_t)(void*)sW;  // LDS byte address of sW

    const int b    = blockIdx.x;
    const int tid  = threadIdx.x;
    const int wid  = tid >> 5;
    const int lane = tid & 31;
    const int lm   = lane & 15;
    const int hiOff = (lane < 16) ? 0 : 8;

    sOut[tid] = 0.f;

    // fact[b] (200x128 f32) -> bf16 LDS, pad rows 200..207 with zeros
    {
        const float4* src = (const float4*)(fact + (size_t)b * LSZ * DFf);
        for (int i = tid; i < (LSZ * DFf) / 4; i += 256) {
            float4 v = src[i];
            int l = (i * 4) / DFf, d = (i * 4) % DFf;
            bf16* dst = sFact + l * STRD + d;
            dst[0] = (bf16)v.x; dst[1] = (bf16)v.y; dst[2] = (bf16)v.z; dst[3] = (bf16)v.w;
        }
        for (int i = tid; i < 8 * DFf; i += 256) {
            int l = LSZ + i / DFf, d = i % DFf;
            sFact[l * STRD + d] = (bf16)0.f;
        }
    }
    __syncthreads();

    for (int h = 0; h < HH; ++h) {
        // ---- F path, layer 1 ----
        stage_weights_async(Wf1t + (size_t)h * 128 * 128, sWlds, 128, tid);
        if (tid < 128) sB[tid] = bf1[h * 128 + tid];
        wait_async_all();
        __syncthreads();
        gemm_store_hid<8>(sFact, sW, sB, sHid, wid, lane);
        __syncthreads();

        // ---- F path, layer 2 + dot with fq ----
        stage_weights_async(Wf2t + (size_t)h * 64 * 128, sWlds, 64, tid);
        if (tid < 64) sB[tid] = bf2[h * 64 + tid];
        if (tid < LP) sDot[tid] = 0.f;
        wait_async_all();
        __syncthreads();

        {
            const float* fqp = fqv + ((size_t)h * BSZ + b) * DD2;
            float fqL[4];
#pragma unroll
            for (int nt = 0; nt < 4; ++nt) fqL[nt] = fqp[nt * 16 + lm];

            for (int mt = wid; mt < 13; mt += 8) {
                const int mBase = mt * 16;
                v8f acc[4];
#pragma unroll
                for (int nt = 0; nt < 4; ++nt)
#pragma unroll
                    for (int r = 0; r < 8; ++r) acc[nt][r] = 0.f;
#pragma unroll
                for (int ks = 0; ks < 4; ++ks) {
                    v16bf a = load_a(sHid, mBase, ks * 32, lane);
#pragma unroll
                    for (int nt = 0; nt < 4; ++nt) {
                        v16bf bb = load_b(sW, nt * 16, ks * 32, lane);
                        acc[nt] = wmma_bf16(a, bb, acc[nt]);
                    }
                }
                float dp[8];
#pragma unroll
                for (int r = 0; r < 8; ++r) dp[r] = 0.f;
#pragma unroll
                for (int nt = 0; nt < 4; ++nt) {
                    const float bias = sB[nt * 16 + lm];
                    const float fql = fqL[nt];
#pragma unroll
                    for (int r = 0; r < 8; ++r) dp[r] += silu_f(acc[nt][r] + bias) * fql;
                }
#pragma unroll
                for (int r = 0; r < 8; ++r) {
                    float v = dp[r];
                    v += __shfl_xor(v, 8, 32);
                    v += __shfl_xor(v, 4, 32);
                    v += __shfl_xor(v, 2, 32);
                    v += __shfl_xor(v, 1, 32);
                    const int row = mBase + r + hiOff;
                    if (lm == 0 && row < LSZ) atomicAdd(&sDot[row], v);
                }
            }
        }
        __syncthreads();

        // ---- mask + cosine bias + softmax over L ----
        {
            const int l = tid;
            float val = -3.0e38f;
            if (l < LSZ) {
                float raw = sDot[l];
                raw += -1e9f * (1.f - (float)mask[b * LSZ + l]);
                float bias = mm[b * LSZ + l] / tau1[0 * HH + h] +
                             mm[(size_t)BSZ * LSZ + b * LSZ + l] / tau1[1 * HH + h];
                val = tau2[0 * HH + h] * raw + tau2[1 * HH + h] * bias +
                      tau2[2 * HH + h] * raw * bias;
            }
            float m = val;
            for (int off = 16; off; off >>= 1) m = fmaxf(m, __shfl_xor(m, off, 32));
            if (lane == 0) sRed[wid] = m;
            __syncthreads();
            if (tid == 0) {
                float mx = sRed[0];
                for (int i = 1; i < 8; ++i) mx = fmaxf(mx, sRed[i]);
                sRed[8] = mx;
            }
            __syncthreads();
            const float mx = sRed[8];
            float e = (l < LSZ) ? __expf(val - mx) : 0.f;
            float s = e;
            for (int off = 16; off; off >>= 1) s += __shfl_xor(s, off, 32);
            if (lane == 0) sRed[wid] = s;
            __syncthreads();
            if (tid == 0) {
                float ss = 0.f;
                for (int i = 0; i < 8; ++i) ss += sRed[i];
                sRed[9] = ss;
            }
            __syncthreads();
            const float denom = sRed[9];
            if (l < LP) sDot[l] = (l < LSZ) ? (e / denom + 1e-7f) : 0.f;
        }
        __syncthreads();

        // ---- V path, layer 1 ----
        stage_weights_async(Wv1t + (size_t)h * 128 * 128, sWlds, 128, tid);
        if (tid < 128) sB[tid] = bv1[h * 128 + tid];
        wait_async_all();
        __syncthreads();
        gemm_store_hid<8>(sFact, sW, sB, sHid, wid, lane);
        __syncthreads();

        // ---- V path, layer 2 + alpha-weighted reduce into sOut ----
        stage_weights_async(Wv2t + (size_t)h * 64 * 128, sWlds, 64, tid);
        if (tid < 64) sB[tid] = bv2[h * 64 + tid];
        wait_async_all();
        __syncthreads();

        for (int mt = wid; mt < 13; mt += 8) {
            const int mBase = mt * 16;
            v8f acc[4];
#pragma unroll
            for (int nt = 0; nt < 4; ++nt)
#pragma unroll
                for (int r = 0; r < 8; ++r) acc[nt][r] = 0.f;
#pragma unroll
            for (int ks = 0; ks < 4; ++ks) {
                v16bf a = load_a(sHid, mBase, ks * 32, lane);
#pragma unroll
                for (int nt = 0; nt < 4; ++nt) {
                    v16bf bb = load_b(sW, nt * 16, ks * 32, lane);
                    acc[nt] = wmma_bf16(a, bb, acc[nt]);
                }
            }
#pragma unroll
            for (int nt = 0; nt < 4; ++nt) {
                const int col = nt * 16 + lm;
                const float bias = sB[col];
                float o = 0.f;
#pragma unroll
                for (int r = 0; r < 8; ++r) {
                    const int row = mBase + r + hiOff;
                    o += silu_f(acc[nt][r] + bias) * sDot[row];
                }
                atomicAdd(&sOut[h * 64 + col], o);
            }
        }
        __syncthreads();
    }

    out[(size_t)b * (HH * DD2) + tid] = sOut[tid];
}

extern "C" void kernel_launch(void* const* d_in, const int* in_sizes, int n_in,
                              void* d_out, int out_size, void* d_ws, size_t ws_size,
                              hipStream_t stream) {
    (void)in_sizes; (void)n_in; (void)out_size; (void)ws_size;
    const float* query = (const float*)d_in[0];
    const float* fact  = (const float*)d_in[1];
    const int*   mask  = (const int*)d_in[2];
    const float* mm    = (const float*)d_in[3];
    const float* Wq1   = (const float*)d_in[4];
    const float* bq1   = (const float*)d_in[5];
    const float* Wq2   = (const float*)d_in[6];
    const float* bq2   = (const float*)d_in[7];
    const float* Wf1   = (const float*)d_in[8];
    const float* bf1   = (const float*)d_in[9];
    const float* Wf2   = (const float*)d_in[10];
    const float* bf2   = (const float*)d_in[11];
    const float* Wv1   = (const float*)d_in[12];
    const float* bv1   = (const float*)d_in[13];
    const float* Wv2   = (const float*)d_in[14];
    const float* bv2   = (const float*)d_in[15];
    const float* tau1  = (const float*)d_in[16];
    const float* tau2  = (const float*)d_in[17];
    float* out = (float*)d_out;

    char* ws = (char*)d_ws;
    float* fqvP  = (float*)(ws + WS_FQV);
    float* hidQ  = (float*)(ws + WS_HIDQ);
    bf16*  Wf1t  = (bf16*)(ws + WS_WF1T);
    bf16*  Wv1t  = (bf16*)(ws + WS_WV1T);
    bf16*  Wf2t  = (bf16*)(ws + WS_WF2T);
    bf16*  Wv2t  = (bf16*)(ws + WS_WV2T);

    // weight transpose+convert to bf16 [H][OUT][IN]
    k_transpose<<<256, 256, 0, stream>>>(Wf1, Wf1t, DFf, DD1);
    k_transpose<<<256, 256, 0, stream>>>(Wv1, Wv1t, DFf, DD1);
    k_transpose<<<128, 256, 0, stream>>>(Wf2, Wf2t, DD1, DD2);
    k_transpose<<<128, 256, 0, stream>>>(Wv2, Wv2t, DD1, DD2);

    // fq pipeline (tiny)
    k_hidq<<<(HH * BSZ * DD1 + 255) / 256, 256, 0, stream>>>(query, Wq1, bq1, hidQ);
    k_fq<<<(HH * BSZ * DD2 + 255) / 256, 256, 0, stream>>>(hidQ, Wq2, bq2, fqvP);

    // main fused attention kernel, one workgroup per batch element
    k_attn<<<BSZ, 256, SM_TOTAL, stream>>>(fact, mask, mm, bf1, bf2, bv1, bv2,
                                           tau1, tau2, fqvP, Wf1t, Wf2t, Wv1t, Wv2t, out);
}
